// Model_31318901523072
// MI455X (gfx1250) — matrix operand order repaired and checked by reading, not compiled, not run on hardware
//
#include <hip/hip_runtime.h>
#include <hip/hip_bf16.h>

// ---------------- problem constants ----------------
#define T_STEPS 300
#define BATCH   256
#define NIN     128
#define NH      512
#define NOUT    3
#define KTOT    (NIN + NH)          // 640 : concatenated K dimension
#define DT_SEC  0.01f
#define ALPHA_N 0.2f

typedef __attribute__((ext_vector_type(16))) _Float16 v16h;
typedef __attribute__((ext_vector_type(8)))  _Float16 v8h;
typedef __attribute__((ext_vector_type(8)))  float    v8f;

// =====================================================================
// Kernel 1: weight preparation.
//  wcat[n][k]  (f16, K-contiguous, i.e. B-matrix transposed) =
//     k <  NIN : w_in[k][n]  * w_in_mask[k][n]
//     k >= NIN : ei[kk] * relu(w_rnn_base[kk][n] * conn_mask[kk][n]),  kk=k-NIN
//  wouteff[n][o] = relu(w_out[n][o] * w_out_mask[n][o])
// =====================================================================
__global__ __launch_bounds__(256) void prep_weights(
    const float* __restrict__ w_in, const float* __restrict__ w_in_mask,
    const float* __restrict__ w_rnn_base, const float* __restrict__ conn_mask,
    const float* __restrict__ ei_matrix,
    const float* __restrict__ w_out, const float* __restrict__ w_out_mask,
    _Float16* __restrict__ wcat, float* __restrict__ wouteff)
{
    const int stride = gridDim.x * blockDim.x;
    int gtid = blockIdx.x * blockDim.x + threadIdx.x;
    for (int e = gtid; e < NH * KTOT; e += stride) {
        int n = e / KTOT, k = e % KTOT;
        float v;
        if (k < NIN) {
            v = w_in[k * NH + n] * w_in_mask[k * NH + n];
        } else {
            int kk = k - NIN;
            float ei = ei_matrix[kk * NH + kk];   // diagonal +-1
            v = ei * fmaxf(w_rnn_base[kk * NH + n] * conn_mask[kk * NH + n], 0.0f);
        }
        wcat[e] = (_Float16)v;
    }
    for (int e = gtid; e < NH * NOUT; e += stride)
        wouteff[e] = fmaxf(w_out[e] * w_out_mask[e], 0.0f);
}

// =====================================================================
// Kernel 2: step-0 elementwise (STP update + activation pack + input cvt)
// =====================================================================
__global__ __launch_bounds__(256) void init_step(
    const float* __restrict__ x_init, const float* __restrict__ syn_x_init,
    const float* __restrict__ syn_u_init, const float* __restrict__ input_data,
    const float* __restrict__ alpha_std, const float* __restrict__ alpha_stf,
    const float* __restrict__ Uv, const float* __restrict__ dyn,
    _Float16* __restrict__ A0, float* __restrict__ x_post,
    float* __restrict__ syn_x, float* __restrict__ syn_u)
{
    const int stride = gridDim.x * blockDim.x;
    int gtid = blockIdx.x * blockDim.x + threadIdx.x;
    for (int i = gtid; i < BATCH * NH; i += stride) {
        int b = i >> 9, n = i & (NH - 1);
        float x0 = x_init[i];
        float r0 = fmaxf(x0, 0.0f);
        float sx = syn_x_init[i], su = syn_u_init[i];
        float d = dyn[n], u = Uv[n];
        float sxn = sx + (alpha_std[n] * (1.0f - sx) - DT_SEC * su * sx * r0) * d;
        float sun = su + (alpha_stf[n] * (u - su) + DT_SEC * u * (1.0f - su) * r0) * d;
        sxn = fminf(fmaxf(sxn, 0.0f), 1.0f);
        sun = fminf(fmaxf(sun, 0.0f), 1.0f);
        syn_x[i] = sxn; syn_u[i] = sun;
        float xp = sun * sxn * x0;
        x_post[i] = xp;
        A0[b * KTOT + NIN + n] = (_Float16)fmaxf(xp, 0.0f);
    }
    for (int i = gtid; i < BATCH * NIN; i += stride) {
        int b = i >> 7, k = i & (NIN - 1);
        A0[b * KTOT + k] = (_Float16)input_data[i];   // t = 0 slice
    }
}

// =====================================================================
// Kernel 3: one recurrent step.
//   C[256x512] = A[256x640](f16) x Wt[640x512](f16), f32 accum via
//   v_wmma_f32_16x16x32_f16. Epilogue: x_new, write x_seq, then the
//   NEXT step's elementwise STP update + f16 activation pack (ping-pong).
//   Grid (4,4): blockIdx.x -> 64-row M tile, blockIdx.y -> 128-col N tile.
//   8 waves: wave = wm(4) x wn(2); each wave does a 16x64 strip (4 WMMAs/K-step).
// =====================================================================
__global__ __launch_bounds__(256) void rnn_step(
    const _Float16* __restrict__ Acur, _Float16* __restrict__ Anext,
    const _Float16* __restrict__ Wt,
    float* __restrict__ x_post, float* __restrict__ syn_x, float* __restrict__ syn_u,
    const float* __restrict__ input_data, const float* __restrict__ noise,
    const float* __restrict__ b_rnn, const float* __restrict__ alpha_std,
    const float* __restrict__ alpha_stf, const float* __restrict__ Uv,
    const float* __restrict__ dyn, float* __restrict__ x_seq,
    int t, int hasNext)
{
    const int lane = threadIdx.x & 31;
    const int wave = threadIdx.x >> 5;
    const int wm = wave & 3;        // 4 waves along M
    const int wn = wave >> 2;       // 2 waves along N
    const int m0 = blockIdx.x * 64 + wm * 16;
    const int n0 = blockIdx.y * 128 + wn * 64;
    const int hs  = lane >> 4;      // lane half-select
    const int l15 = lane & 15;

    v8f acc[4];
    #pragma unroll
    for (int j = 0; j < 4; ++j) acc[j] = (v8f){0.f,0.f,0.f,0.f,0.f,0.f,0.f,0.f};

    // A fragment: lane holds row (m0+l15); 16 contiguous K values {hs*16..hs*16+15}
    const _Float16* arow = Acur + (size_t)(m0 + l15) * KTOT + hs * 16;
    // B fragment: lane holds column (n0+j*16+l15); 16 contiguous K values
    const _Float16* bcol = Wt + (size_t)(n0 + l15) * KTOT + hs * 16;

    for (int kb = 0; kb < KTOT; kb += 32) {
        union { v16h v; v8h h[2]; } a;
        a.h[0] = *(const v8h*)(arow + kb);
        a.h[1] = *(const v8h*)(arow + kb + 8);
        v16h b0 = *(const v16h*)(bcol + kb);
        v16h b1 = *(const v16h*)(bcol + 16 * KTOT + kb);
        v16h b2 = *(const v16h*)(bcol + 32 * KTOT + kb);
        v16h b3 = *(const v16h*)(bcol + 48 * KTOT + kb);
        acc[0] = __builtin_amdgcn_wmma_f32_16x16x32_f16(false, a.v, false, b0, (short)0, acc[0], false, false);
        acc[1] = __builtin_amdgcn_wmma_f32_16x16x32_f16(false, a.v, false, b1, (short)0, acc[1], false, false);
        acc[2] = __builtin_amdgcn_wmma_f32_16x16x32_f16(false, a.v, false, b2, (short)0, acc[2], false, false);
        acc[3] = __builtin_amdgcn_wmma_f32_16x16x32_f16(false, a.v, false, b3, (short)0, acc[3], false, false);
    }

    // ----- epilogue: x_new + fused next-step STP update -----
    const float* noise_t = noise + (size_t)t * (BATCH * NH);
    #pragma unroll
    for (int j = 0; j < 4; ++j) {
        const int n = n0 + j * 16 + l15;
        const float brn = b_rnn[n];
        const float astd = alpha_std[n], astf = alpha_stf[n];
        const float u = Uv[n], d = dyn[n];
        #pragma unroll
        for (int r = 0; r < 8; ++r) {
            const int brow = m0 + r + hs * 8;
            const int idx = brow * NH + n;
            float xp = x_post[idx];
            float xnew = (1.0f - ALPHA_N) * xp
                       + ALPHA_N * (acc[j][r] + brn)
                       + noise_t[idx];
            // x_seq[b][t][n], b-major
            x_seq[(size_t)brow * (T_STEPS * NH) + (size_t)t * NH + n] = xnew;
            if (hasNext) {
                float rr = fmaxf(xnew, 0.0f);
                float sx = syn_x[idx], su = syn_u[idx];
                float sxn = sx + (astd * (1.0f - sx) - DT_SEC * su * sx * rr) * d;
                float sun = su + (astf * (u - su) + DT_SEC * u * (1.0f - su) * rr) * d;
                sxn = fminf(fmaxf(sxn, 0.0f), 1.0f);
                sun = fminf(fmaxf(sun, 0.0f), 1.0f);
                syn_x[idx] = sxn; syn_u[idx] = sun;
                float xpn = sun * sxn * xnew;
                x_post[idx] = xpn;
                Anext[(size_t)brow * KTOT + NIN + n] = (_Float16)fmaxf(xpn, 0.0f);
            }
        }
    }

    // ----- convert next timestep's input slice (first N-column blocks only) -----
    if (hasNext && blockIdx.y == 0) {
        const int base_b = blockIdx.x * 64;
        const float* src = input_data + (size_t)(t + 1) * (BATCH * NIN);
        for (int i = threadIdx.x; i < 64 * NIN; i += 256) {
            int rr = i >> 7, cc = i & (NIN - 1);
            int b = base_b + rr;
            Anext[(size_t)b * KTOT + cc] = (_Float16)src[b * NIN + cc];
        }
    }
}

// =====================================================================
// Kernel 4: output projection y_raw[b][t][o] = relu(x_seq[b][t][:]) @ wouteff + b_out
//   one wave per (b,t) row, lane-strided dot, cross-lane reduce.
// =====================================================================
__global__ __launch_bounds__(256) void y_project(
    const float* __restrict__ x_seq, const float* __restrict__ wouteff,
    const float* __restrict__ b_out, float* __restrict__ y_raw)
{
    int gwave = (blockIdx.x * 256 + threadIdx.x) >> 5;   // 0 .. B*T-1
    int lane  = threadIdx.x & 31;
    if (gwave >= BATCH * T_STEPS) return;
    const float* xr = x_seq + (size_t)gwave * NH;
    float a0 = 0.f, a1 = 0.f, a2 = 0.f;
    for (int k = lane; k < NH; k += 32) {
        float v = fmaxf(xr[k], 0.0f);
        a0 += v * wouteff[k * NOUT + 0];
        a1 += v * wouteff[k * NOUT + 1];
        a2 += v * wouteff[k * NOUT + 2];
    }
    #pragma unroll
    for (int off = 16; off > 0; off >>= 1) {
        a0 += __shfl_xor(a0, off, 32);
        a1 += __shfl_xor(a1, off, 32);
        a2 += __shfl_xor(a2, off, 32);
    }
    if (lane == 0) {
        y_raw[gwave * NOUT + 0] = a0 + b_out[0];
        y_raw[gwave * NOUT + 1] = a1 + b_out[1];
        y_raw[gwave * NOUT + 2] = a2 + b_out[2];
    }
}

// =====================================================================
// Kernel 5: softmax over the batch axis (axis 0 of [B,T,NOUT]).
//   one block per (t,o) column of 256 batch entries.
// =====================================================================
__global__ __launch_bounds__(256) void softmax_batch(
    const float* __restrict__ y_raw, float* __restrict__ yout)
{
    __shared__ float sm[256];
    const int t = blockIdx.x / NOUT;
    const int o = blockIdx.x % NOUT;
    const int b = threadIdx.x;
    const int idx = (b * T_STEPS + t) * NOUT + o;
    float v = y_raw[idx];
    sm[b] = v; __syncthreads();
    for (int s = 128; s > 0; s >>= 1) {
        if (b < s) sm[b] = fmaxf(sm[b], sm[b + s]);
        __syncthreads();
    }
    float mx = sm[0]; __syncthreads();
    float e = __expf(v - mx);
    sm[b] = e; __syncthreads();
    for (int s = 128; s > 0; s >>= 1) {
        if (b < s) sm[b] += sm[b + s];
        __syncthreads();
    }
    yout[idx] = e / sm[0];
}

// =====================================================================
// host launcher
// =====================================================================
extern "C" void kernel_launch(void* const* d_in, const int* in_sizes, int n_in,
                              void* d_out, int out_size, void* d_ws, size_t ws_size,
                              hipStream_t stream) {
    (void)in_sizes; (void)n_in; (void)out_size; (void)ws_size;
    const float* input_data = (const float*)d_in[0];
    const float* noise      = (const float*)d_in[1];
    const float* x_init     = (const float*)d_in[2];
    const float* syn_x_init = (const float*)d_in[3];
    const float* syn_u_init = (const float*)d_in[4];
    const float* w_in       = (const float*)d_in[5];
    const float* w_in_mask  = (const float*)d_in[6];
    const float* w_rnn_base = (const float*)d_in[7];
    const float* conn_mask  = (const float*)d_in[8];
    const float* ei_matrix  = (const float*)d_in[9];
    const float* b_rnn      = (const float*)d_in[10];
    const float* w_out      = (const float*)d_in[11];
    const float* w_out_mask = (const float*)d_in[12];
    const float* b_out      = (const float*)d_in[13];
    const float* alpha_std  = (const float*)d_in[14];
    const float* alpha_stf  = (const float*)d_in[15];
    const float* U          = (const float*)d_in[16];
    const float* dyn        = (const float*)d_in[17];

    // workspace layout (256B-aligned slabs, ~3.7 MB total)
    char* ws = (char*)d_ws;
    _Float16* wcat    = (_Float16*)(ws);                 // 512*640*2   = 655360
    float*    wouteff = (float*)   (ws + 655360);        // 512*3*4     = 6144
    _Float16* A0      = (_Float16*)(ws + 661504);        // 256*640*2   = 327680
    _Float16* A1      = (_Float16*)(ws + 989184);        // 256*640*2   = 327680
    float*    x_post  = (float*)   (ws + 1316864);       // 256*512*4   = 524288
    float*    syn_x   = (float*)   (ws + 1841152);       // 524288
    float*    syn_u   = (float*)   (ws + 2365440);       // 524288
    float*    y_raw   = (float*)   (ws + 2889728);       // 256*300*3*4 = 921600

    float* yout = (float*)d_out;                          // [B,T,NOUT]
    float* xout = yout + (size_t)BATCH * T_STEPS * NOUT;  // [B,T,NH]

    prep_weights<<<256, 256, 0, stream>>>(w_in, w_in_mask, w_rnn_base, conn_mask,
                                          ei_matrix, w_out, w_out_mask, wcat, wouteff);
    init_step<<<512, 256, 0, stream>>>(x_init, syn_x_init, syn_u_init, input_data,
                                       alpha_std, alpha_stf, U, dyn,
                                       A0, x_post, syn_x, syn_u);
    for (int t = 0; t < T_STEPS; ++t) {
        const _Float16* Ac = (t & 1) ? A1 : A0;
        _Float16*       An = (t & 1) ? A0 : A1;
        rnn_step<<<dim3(4, 4), 256, 0, stream>>>(Ac, An, wcat, x_post, syn_x, syn_u,
                                                 input_data, noise, b_rnn,
                                                 alpha_std, alpha_stf, U, dyn,
                                                 xout, t, (t + 1 < T_STEPS) ? 1 : 0);
    }
    y_project<<<(BATCH * T_STEPS) / 8, 256, 0, stream>>>(xout, wouteff, b_out, y_raw);
    softmax_batch<<<T_STEPS * NOUT, 256, 0, stream>>>(y_raw, yout);
}
